// WTA_19602230739131
// MI455X (gfx1250) — matrix operand order repaired and checked by reading, not compiled, not run on hardware
//
#include <hip/hip_runtime.h>
#include <stdint.h>

typedef __attribute__((ext_vector_type(16))) _Float16 v16h;
typedef __attribute__((ext_vector_type(8)))  _Float16 v8h;
typedef __attribute__((ext_vector_type(2)))  _Float16 v2h;
typedef __attribute__((ext_vector_type(8)))  float    v8f;
typedef __attribute__((ext_vector_type(4)))  float    f4;

#define THRV   0.03f
#define DECAYV 0.951229424500714f     /* exp(-1/20) */
#define OMDV   (1.0f - DECAYV)

// Fused  out = A0 @ W0 - A1 @ W1  followed by the LIF neuron update.
// A0:[8192,K0], W0:[K0,NCOLS], A1:[8192,K1], W1:[K1,NCOLS].
template<int NCOLS, int K0, int K1>
__global__ __launch_bounds__(256)
void snn_fused(const float* __restrict__ A0, const float* __restrict__ W0,
               const float* __restrict__ A1, const float* __restrict__ W1,
               const float* __restrict__ old_v, const float* __restrict__ old_spike,
               const float* __restrict__ last_dist, float ref_period,
               float* __restrict__ z_out, float* __restrict__ vsc_out,
               float* __restrict__ newv_out, float* __restrict__ newlast_out)
{
    constexpr int BM = 128, BN = 64, BK = 32;
    constexpr int AP = 40;               // A tile pitch in floats (160B rows, 16B aligned)
    constexpr int BP = 40;               // B tile pitch in halfs  (80B rows, 16B aligned)
    constexpr int ABYTES = BM * AP * 4;  // 20480
    constexpr int BBYTES = BN * BP * 2;  // 5120
    constexpr int CP = 68;               // C repack pitch in floats (272B rows, 16B aligned)
    constexpr int S  = (K0 + K1) / BK;

    __shared__ __align__(16) unsigned char smem[2 * ABYTES + 2 * BBYTES]; // 51200B

    const int tid  = threadIdx.x;
    const int lane = tid & 31;
    const int wave = tid >> 5;
    const int mBase = blockIdx.x * BM;
    const int nBase = blockIdx.y * BN;

    auto Abuf = [&](int b) -> float*     { return (float*)(smem + b * ABYTES); };
    auto Bbuf = [&](int b) -> _Float16*  { return (_Float16*)(smem + 2 * ABYTES + b * BBYTES); };
    float* Cs = (float*)smem;            // reused after the K loop

    // ---- A tile: async DMA global f32 -> LDS (128 x 32 f32). 1024 16B chunks,
    //      8 waves x 4 issues x 32 lanes. Tracked by ASYNCcnt.
    auto stageA = [&](int b, int kt) {
        const float* Asrc; int lda, kloc;
        if (kt < K0) { Asrc = A0; lda = K0; kloc = kt; }
        else         { Asrc = A1; lda = K1; kloc = kt - K0; }
        const unsigned ldsBase = (unsigned)(size_t)Abuf(b);
        #pragma unroll
        for (int t = 0; t < 4; ++t) {
            const int chunk = ((wave * 4 + t) << 5) | lane;   // 0..1023
            const int row = chunk >> 3;                       // 0..127
            const int kb  = chunk & 7;                        // 16B chunk in row
            const unsigned lds = ldsBase + row * (AP * 4) + kb * 16;
            const unsigned long long ga =
                (unsigned long long)(size_t)(Asrc + (size_t)(mBase + row) * lda + kloc + kb * 4);
            asm volatile("global_load_async_to_lds_b128 %0, %1, off"
                         :: "v"(lds), "v"(ga) : "memory");
        }
    };

    // ---- B tile: 32 x 64 f32 row-major -> LDS f16 column-major [n][k], sign folded.
    //      Each thread: two float4 loads (rows k, k+1), four packed half2 stores.
    auto stageB = [&](int b, int kt) {
        const float* Wsrc = (kt < K0) ? W0 : W1;
        const float  sgn  = (kt < K0) ? 1.0f : -1.0f;
        const int    kloc = (kt < K0) ? kt : kt - K0;
        _Float16* Bb = Bbuf(b);
        const int kp = (tid >> 4) * 2;       // 0,2,..,30
        const int nc = (tid & 15) * 4;       // 0,4,..,60
        const float* p0 = Wsrc + (size_t)(kloc + kp) * NCOLS + nBase + nc;
        const f4 w0 = *(const f4*)p0;
        const f4 w1 = *(const f4*)(p0 + NCOLS);
        #pragma unroll
        for (int c = 0; c < 4; ++c) {
            v2h p;
            p[0] = (_Float16)(sgn * w0[c]);
            p[1] = (_Float16)(sgn * w1[c]);
            *(v2h*)&Bb[(nc + c) * BP + kp] = p;
        }
    };

    v8f acc[4] = {};   // this wave: rows [mBase+wave*16, +16), cols [nBase, +64)

    stageA(0, 0);
    stageB(0, 0);
    asm volatile("s_wait_asynccnt 0x0" ::: "memory");
    __syncthreads();

    for (int s = 0; s < S; ++s) {
        const int cur = s & 1;
        if (s + 1 < S) {           // prefetch next tile into the other buffer
            stageA(cur ^ 1, (s + 1) * BK);
            stageB(cur ^ 1, (s + 1) * BK);
        }

        // ---- A fragment: f32 LDS -> cvt f16. lanes 0-15: K 0..15;
        //      lanes 16-31: K 16..31 (contiguous, matches B fragments).
        const float* Ab  = Abuf(cur);
        const int arow = wave * 16 + (lane & 15);
        const int koff = (lane >> 4) * 16;
        const float* ap = Ab + arow * AP + koff;
        const f4 a0 = *(const f4*)ap;
        const f4 a1 = *(const f4*)(ap + 4);
        const f4 a2 = *(const f4*)(ap + 8);
        const f4 a3 = *(const f4*)(ap + 12);
        v16h a;
        #pragma unroll
        for (int q = 0; q < 4; ++q) {
            a[q]      = (_Float16)a0[q];
            a[4 + q]  = (_Float16)a1[q];
            a[8 + q]  = (_Float16)a2[q];
            a[12 + q] = (_Float16)a3[q];
        }

        // ---- B fragments: lane holds column n = lane&15; 16 contiguous K per half-wave.
        const _Float16* Bb = Bbuf(cur);
        const int kboff = (lane >> 4) * 16;
        #pragma unroll
        for (int j = 0; j < 4; ++j) {
            const _Float16* bp = Bb + (j * 16 + (lane & 15)) * BP + kboff;
            const v8h blo = *(const v8h*)bp;
            const v8h bhi = *(const v8h*)(bp + 8);
            const v16h bf = __builtin_shufflevector(blo, bhi,
                              0,1,2,3,4,5,6,7,8,9,10,11,12,13,14,15);
            acc[j] = __builtin_amdgcn_wmma_f32_16x16x32_f16(
                         false, a, false, bf, (short)0, acc[j], false, false);
        }

        asm volatile("s_wait_asynccnt 0x0" ::: "memory");
        __syncthreads();
    }

    // ---- repack accumulators into LDS (row-major) for a vectorized epilogue ----
    #pragma unroll
    for (int j = 0; j < 4; ++j) {
        #pragma unroll
        for (int r = 0; r < 8; ++r) {
            const int crow = wave * 16 + r + 8 * (lane >> 4);
            const int ccol = j * 16 + (lane & 15);
            Cs[crow * CP + ccol] = acc[j][r];
        }
    }
    __syncthreads();

    // ---- fused LIF epilogue, float4 everywhere (coalesced b128) ----
    #pragma unroll
    for (int t = 0; t < 8; ++t) {
        const int v   = tid + t * 256;       // 0..2047 float4 chunks of 128x64 tile
        const int row = v >> 4;
        const int c4  = (v & 15) * 4;
        const f4 iin = *(const f4*)&Cs[row * CP + c4];
        const size_t gidx = (size_t)(mBase + row) * NCOLS + nBase + c4;
        const f4 ov = *(const f4*)(old_v + gidx);
        const f4 os = *(const f4*)(old_spike + gidx);
        const f4 ld = *(const f4*)(last_dist + gidx);
        f4 z, vsc, nv, nl;
        #pragma unroll
        for (int c = 0; c < 4; ++c) {
            const float n  = DECAYV * ov[c] + OMDV * iin[c] - THRV * os[c];
            const float vs = (n - THRV) * (1.0f / THRV);
            float zz = (vs > 0.0f) ? 1.0f : 0.0f;
            zz *= (ld[c] >= ref_period) ? 1.0f : 0.0f;
            nv[c]  = n;
            vsc[c] = vs;
            z[c]   = zz;
            nl[c]  = (ld[c] + 1.0f) * (1.0f - zz);
        }
        *(f4*)(z_out + gidx)       = z;
        *(f4*)(vsc_out + gidx)     = vsc;
        *(f4*)(newv_out + gidx)    = nv;
        *(f4*)(newlast_out + gidx) = nl;
    }
}

extern "C" void kernel_launch(void* const* d_in, const int* in_sizes, int n_in,
                              void* d_out, int out_size, void* d_ws, size_t ws_size,
                              hipStream_t stream) {
    (void)in_sizes; (void)n_in; (void)out_size; (void)d_ws; (void)ws_size;
    const float* inputs = (const float*)d_in[0];
    const float* osp_i  = (const float*)d_in[1];
    const float* osp_e  = (const float*)d_in[2];
    const float* ov_i   = (const float*)d_in[3];
    const float* ov_e   = (const float*)d_in[4];
    const float* last_i = (const float*)d_in[5];
    const float* last_e = (const float*)d_in[6];
    const float* w_in2e = (const float*)d_in[7];
    const float* w_i2i  = (const float*)d_in[8];
    const float* w_i2e  = (const float*)d_in[9];
    const float* w_e2i  = (const float*)d_in[10];

    float* out = (float*)d_out;
    const size_t BE = (size_t)8192 * 2048;
    const size_t BI = (size_t)8192 * 512;
    // return order: z_e, z_i, v_sc_e, v_sc_i, new_v_i, new_v_e, new_last_i, new_last_e
    float* z_e   = out;
    float* z_i   = out + BE;
    float* vsc_e = out + BE + BI;
    float* vsc_i = out + 2 * BE + BI;
    float* nv_i  = out + 2 * BE + 2 * BI;
    float* nv_e  = out + 2 * BE + 3 * BI;
    float* nl_i  = out + 3 * BE + 3 * BI;
    float* nl_e  = out + 3 * BE + 4 * BI;

    dim3 blk(256);
    // excitatory: i_in_e = inputs @ w_input2exc - old_spike_i @ w_i2e
    snn_fused<2048, 1024, 512><<<dim3(8192 / 128, 2048 / 64), blk, 0, stream>>>(
        inputs, w_in2e, osp_i, w_i2e, ov_e, osp_e, last_e, 5.0f,
        z_e, vsc_e, nv_e, nl_e);
    // inhibitory: i_in_i = old_spike_e @ w_e2i - old_spike_i @ w_i2i
    snn_fused<512, 2048, 512><<<dim3(8192 / 128, 512 / 64), blk, 0, stream>>>(
        osp_e, w_e2i, osp_i, w_i2i, ov_i, osp_i, last_i, 2.0f,
        z_i, vsc_i, nv_i, nl_i);
}
